// STGNN_GhostFusor_AR_87471303950941
// MI455X (gfx1250) — compile-verified, matrix operand order checked by reading, not run on hardware
//
#include <hip/hip_runtime.h>
#include <hip/hip_bf16.h>

#define HID 128
#define TIN 6
#define TOUT 3

typedef __attribute__((ext_vector_type(16))) _Float16 v16h;
typedef __attribute__((ext_vector_type(8)))  float    v8f;
typedef int i32x4 __attribute__((ext_vector_type(4)));
typedef int i32x8 __attribute__((ext_vector_type(8)));

__device__ __forceinline__ float sigm_(float x) { return 1.0f / (1.0f + __expf(-x)); }

// ---------------------------------------------------------------------------
// WMMA GEMM: C[M x Kout] = A[M x K] @ Bt^T (+ bias), A/Bt f16, C fp32.
// Bt stored transposed [Kout x K] row-major.
// Block = 8 waves = 128 rows x 64 cols. Wave = 16 rows x 64 cols (4 acc tiles,
// A fragment reused 4x). The 64x128-f16 (16KB) weight panel for the block is
// staged into LDS by wave 0 via the Tensor Data Mover (tensor_load_to_lds,
// 1-D descriptor, data_size=8B), synchronized with s_wait_tensorcnt + barrier.
// ---------------------------------------------------------------------------
__global__ void k_gemm_wmma(const _Float16* __restrict__ A,
                            const _Float16* __restrict__ Bt,
                            const float* __restrict__ bias,
                            float* __restrict__ C,
                            int M, int K, int Kout)
{
    __shared__ _Float16 sB[64 * HID];          // 16 KB weight panel (K == HID)
    const int colg = blockIdx.y;               // 64-column group

    // ---- TDM: stage Bt rows [colg*64, colg*64+64) x K into LDS (wave 0) ----
    if (threadIdx.x < 32) {
        unsigned ldsOff =
            (unsigned)(unsigned long long)(__attribute__((address_space(3))) _Float16*)sB;
        unsigned long long ga =
            (unsigned long long)(const void*)(Bt + (size_t)colg * 64 * K);
        int elems = (64 * K * 2) >> 3;         // 8-byte elements (2048 for K=128)
        // D# group0: count=1 | lds_addr | global_addr[31:0] | ga[56:32],type=2
        i32x4 g0;
        g0[0] = 1;
        g0[1] = (int)ldsOff;
        g0[2] = (int)(ga & 0xffffffffull);
        g0[3] = (int)(((ga >> 32) & 0x01ffffffull) | (2u << 30));
        // D# group1: data_size=3(8B); tensor_dim0=elems; tensor_dim1=1;
        //            tile_dim0=elems (1-D); tensor_dim0_stride=elems
        i32x8 g1;
        g1[0] = 3 << 16;
        g1[1] = (elems & 0xffff) << 16;        // tensor_dim0[15:0]
        g1[2] = 1 << 16;                       // tensor_dim0[31:16]=0, tensor_dim1=1
        g1[3] = (elems & 0xffff) << 16;        // tile_dim0
        g1[4] = 0;                             // tile_dim1=0 (unused), tile_dim2=0
        g1[5] = elems;                         // tensor_dim0_stride[31:0]
        g1[6] = 0;
        g1[7] = 0;
        asm volatile("tensor_load_to_lds %0, %1" :: "s"(g0), "s"(g1) : "memory");
        __builtin_amdgcn_s_wait_tensorcnt(0);
    }
    __syncthreads();

    const int wid    = threadIdx.x >> 5;
    const int tm     = blockIdx.x * 8 + wid;
    const int tilesM = M >> 4;
    if (tm >= tilesM) return;                  // wave-uniform, after the barrier

    const int lane  = threadIdx.x & 31;
    const int half_ = lane >> 4;
    const int idx   = lane & 15;

    const _Float16* arow = A + (size_t)(tm * 16 + idx) * K;

    v8f acc[4];
#pragma unroll
    for (int c = 0; c < 4; ++c) {
        float bv = bias ? bias[colg * 64 + c * 16 + idx] : 0.0f;
#pragma unroll
        for (int v = 0; v < 8; ++v) acc[c][v] = bv;
    }

    for (int kk = 0; kk < K; kk += 32) {
        __builtin_prefetch(arow + kk + 32, 0, 0);       // global_prefetch_b8
        union { v16h v; unsigned u[8]; } ua;
#pragma unroll
        for (int r = 0; r < 8; ++r) {
            // ISA 7.12.2 f16 A 16x32 lane layout: vgpr r<4 -> K=2r(+8*half),
            // r>=4 -> K=16+2(r-4)(+8*half)
            int kb = ((r < 4) ? (2 * r) : (16 + 2 * (r - 4))) + 8 * half_;
            ua.u[r] = *(const unsigned*)(arow + kk + kb);
        }
#pragma unroll
        for (int c = 0; c < 4; ++c) {
            const _Float16* brow = sB + (size_t)(c * 16 + idx) * K + kk;
            union { v16h v; unsigned u[8]; } ub;
#pragma unroll
            for (int r = 0; r < 8; ++r) {
                int kb = ((r < 4) ? (2 * r) : (16 + 2 * (r - 4))) + 8 * half_;
                ub.u[r] = *(const unsigned*)(brow + kb);   // ds_load from LDS
            }
            acc[c] = __builtin_amdgcn_wmma_f32_16x16x32_f16(
                         false, ua.v, false, ub.v, (short)0, acc[c], false, false);
        }
    }

    // C/D layout: lane idx = column; vgpr v = row v + 8*half
#pragma unroll
    for (int c = 0; c < 4; ++c) {
        float* crow = C + (size_t)(tm * 16) * Kout + colg * 64 + c * 16 + idx;
#pragma unroll
        for (int v = 0; v < 8; ++v)
            crow[(size_t)(v + 8 * half_) * Kout] = acc[c][v];
    }
}

// --------------------------- prep kernels ----------------------------------
__global__ void k_init_nd(int* __restrict__ me, float* __restrict__ deg, int Nn)
{
    int n = blockIdx.x * blockDim.x + threadIdx.x;
    if (n >= Nn) return;
    me[n]  = 0x7fffffff;
    deg[n] = 1.0f;                // self loop
}

__global__ void k_edge_prep(const int* __restrict__ src, const int* __restrict__ dst,
                            int* __restrict__ me, float* __restrict__ deg, int E)
{
    int e = blockIdx.x * blockDim.x + threadIdx.x;
    if (e >= E) return;
    atomicMin(&me[src[e]], e);
    atomicAdd(&deg[dst[e]], 1.0f);
}

__global__ void k_node_prep(const int* __restrict__ me, const int* __restrict__ dst,
                            const float* __restrict__ eattr, const float* __restrict__ deg,
                            int* __restrict__ fo, float* __restrict__ fdx,
                            float* __restrict__ dinv, int Nn, int E)
{
    int n = blockIdx.x * blockDim.x + threadIdx.x;
    if (n >= Nn) return;
    int  m   = me[n];
    bool has = m < E;
    int  mc  = has ? m : (E - 1);
    fo[n]   = has ? dst[mc] : -1;
    fdx[n]  = has ? fmaxf(eattr[2 * mc], 1e-6f) : -1.0f;
    dinv[n] = rsqrtf(deg[n]);
}

__global__ void k_copy(const float* __restrict__ a, float* __restrict__ b, int cnt)
{
    int i = blockIdx.x * blockDim.x + threadIdx.x;
    if (i < cnt) b[i] = a[i];
}

// Ghost fusor: tiny MLPs (4->32->1 and 4->16->1), one thread per (ghost, t)
__global__ void k_ghost(const float* __restrict__ x, const int* __restrict__ fo,
                        const float* __restrict__ fdx,
                        const float* __restrict__ w1, const float* __restrict__ b1,
                        const float* __restrict__ w2, const float* __restrict__ b2,
                        const float* __restrict__ gw1, const float* __restrict__ gb1,
                        const float* __restrict__ gw2, const float* __restrict__ gb2,
                        float* __restrict__ xf, int G)
{
    int tid = blockIdx.x * blockDim.x + threadIdx.x;
    if (tid >= G * TIN) return;
    int g = tid / TIN, t = tid % TIN;
    int bn = fo[g];  if (bn < 0) bn = g;
    int d1 = fo[bn]; if (d1 < 0) d1 = bn;
    int d2 = fo[d1]; if (d2 < 0) d2 = d1;
    float inp[4];
    inp[0] = x[(size_t)g  * TIN + t];
    inp[1] = x[(size_t)d1 * TIN + t];
    inp[2] = x[(size_t)d2 * TIN + t];
    inp[3] = fmaxf(fdx[g], 1e-6f);
    float delta = b2[0];
#pragma unroll 4
    for (int j = 0; j < 32; ++j) {
        float z = b1[j];
        for (int i = 0; i < 4; ++i) z += inp[i] * w1[i * 32 + j];
        delta += fmaxf(z, 0.0f) * w2[j];
    }
    float gz = gb2[0];
#pragma unroll 4
    for (int j = 0; j < 16; ++j) {
        float z = gb1[j];
        for (int i = 0; i < 4; ++i) z += inp[i] * gw1[i * 16 + j];
        gz += fmaxf(z, 0.0f) * gw2[j];
    }
    xf[(size_t)g * TIN + t] = inp[0] + 0.5f * delta * sigm_(gz);
}

// fp32 weight [K x Kout] -> f16 transposed [Kout x K]
__global__ void k_w2f16t(const float* __restrict__ W, _Float16* __restrict__ Wt,
                         int K, int Kout)
{
    int tid = blockIdx.x * blockDim.x + threadIdx.x;
    if (tid >= K * Kout) return;
    int i = tid / Kout, o = tid % Kout;
    Wt[(size_t)o * K + i] = (_Float16)W[tid];
}

__global__ void k_zero_h(float* __restrict__ h, _Float16* __restrict__ hf, int cnt)
{
    int i = blockIdx.x * blockDim.x + threadIdx.x;
    if (i >= cnt) return;
    h[i]  = 0.0f;
    hf[i] = (_Float16)0.0f;
}

// ------------------------- per-timestep kernels ----------------------------
__global__ void k_lin_relu(const float* __restrict__ xs, int stride,
                           const float* __restrict__ w, const float* __restrict__ b,
                           _Float16* __restrict__ o, int Nn)
{
    int tid = blockIdx.x * blockDim.x + threadIdx.x;
    if (tid >= Nn * HID) return;
    int n = tid >> 7, j = tid & (HID - 1);
    o[tid] = (_Float16)fmaxf(xs[(size_t)n * stride] * w[j] + b[j], 0.0f);
}

__global__ void k_conv_init(const float* __restrict__ xw, const float* __restrict__ dinv,
                            float* __restrict__ agg, int Nn)
{
    int tid = blockIdx.x * blockDim.x + threadIdx.x;
    if (tid >= Nn * HID) return;
    int n = tid >> 7;
    float w = dinv[n] * dinv[n];
    agg[tid] = xw[tid] * w;
}

// scatter: one wave per edge, 4 features/lane, L2-resident float atomics
__global__ void k_conv_edges(const int* __restrict__ src, const int* __restrict__ dst,
                             const float* __restrict__ dinv, const float* __restrict__ xw,
                             float* __restrict__ agg, int E)
{
    long tid = (long)blockIdx.x * blockDim.x + threadIdx.x;
    int e = (int)(tid >> 5);
    if (e >= E) return;
    int lane = (int)(tid & 31);
    int s = src[e], d = dst[e];
    float w = dinv[s] * dinv[d];
    float4 xv = *(const float4*)(xw + (size_t)s * HID + lane * 4);
    float* ap = agg + (size_t)d * HID + lane * 4;
    atomicAdd(ap + 0, xv.x * w);
    atomicAdd(ap + 1, xv.y * w);
    atomicAdd(ap + 2, xv.z * w);
    atomicAdd(ap + 3, xv.w * w);
}

__global__ void k_conv_fin(const float* __restrict__ agg, const float* __restrict__ bias,
                           _Float16* __restrict__ o, int Nn)
{
    int tid = blockIdx.x * blockDim.x + threadIdx.x;
    if (tid >= Nn * HID) return;
    int j = tid & (HID - 1);
    o[tid] = (_Float16)fmaxf(agg[tid] + bias[j], 0.0f);
}

__global__ void k_gru(const float* __restrict__ gi, const float* __restrict__ gh,
                      float* __restrict__ h, _Float16* __restrict__ hf, int Nn)
{
    int tid = blockIdx.x * blockDim.x + threadIdx.x;
    if (tid >= Nn * HID) return;
    int n = tid >> 7, j = tid & (HID - 1);
    const float* gin = gi + (size_t)n * 3 * HID;
    const float* ghn = gh + (size_t)n * 3 * HID;
    float r  = sigm_(gin[j] + ghn[j]);
    float u  = sigm_(gin[HID + j] + ghn[HID + j]);
    float nn = tanhf(gin[2 * HID + j] + r * ghn[2 * HID + j]);
    float hv = (1.0f - u) * nn + u * h[tid];
    h[tid]  = hv;
    hf[tid] = (_Float16)hv;
}

// LayerNorm(128) + ReLU + f16 cast: one wave per node, shfl reductions (wave32)
__global__ void k_ln_relu_f16(const float* __restrict__ Z, const float* __restrict__ g,
                              const float* __restrict__ b, _Float16* __restrict__ O, int Nn)
{
    int wave = (blockIdx.x * blockDim.x + threadIdx.x) >> 5;
    int lane = threadIdx.x & 31;
    if (wave >= Nn) return;
    const float* z = Z + (size_t)wave * HID;
    float v0 = z[lane], v1 = z[lane + 32], v2 = z[lane + 64], v3 = z[lane + 96];
    float s = v0 + v1 + v2 + v3;
#pragma unroll
    for (int m = 16; m >= 1; m >>= 1) s += __shfl_xor(s, m, 32);
    float mu = s * (1.0f / HID);
    float d0 = v0 - mu, d1 = v1 - mu, d2 = v2 - mu, d3 = v3 - mu;
    float q = d0 * d0 + d1 * d1 + d2 * d2 + d3 * d3;
#pragma unroll
    for (int m = 16; m >= 1; m >>= 1) q += __shfl_xor(q, m, 32);
    float rs = rsqrtf(q * (1.0f / HID) + 1e-5f);
    _Float16* o = O + (size_t)wave * HID;
    o[lane]      = (_Float16)fmaxf(d0 * rs * g[lane]      + b[lane],      0.0f);
    o[lane + 32] = (_Float16)fmaxf(d1 * rs * g[lane + 32] + b[lane + 32], 0.0f);
    o[lane + 64] = (_Float16)fmaxf(d2 * rs * g[lane + 64] + b[lane + 64], 0.0f);
    o[lane + 96] = (_Float16)fmaxf(d3 * rs * g[lane + 96] + b[lane + 96], 0.0f);
}

// head: y = q . w3 + b3 ; feed back into cur, emit real-node outputs
__global__ void k_pred_out(const _Float16* __restrict__ Q, const float* __restrict__ w3,
                           const float* __restrict__ b3, float* __restrict__ cur,
                           float* __restrict__ out, int Nn, int G, int t)
{
    int wave = (blockIdx.x * blockDim.x + threadIdx.x) >> 5;
    int lane = threadIdx.x & 31;
    if (wave >= Nn) return;
    const _Float16* q = Q + (size_t)wave * HID;
    float acc = (float)q[lane]      * w3[lane]
              + (float)q[lane + 32] * w3[lane + 32]
              + (float)q[lane + 64] * w3[lane + 64]
              + (float)q[lane + 96] * w3[lane + 96];
#pragma unroll
    for (int m = 16; m >= 1; m >>= 1) acc += __shfl_xor(acc, m, 32);
    if (lane == 0) {
        float y = acc + b3[0];
        cur[wave] = y;
        if (wave >= G) out[(size_t)(wave - G) * TOUT + t] = y;
    }
}

__global__ void k_set_cur(const float* __restrict__ xf, float* __restrict__ cur, int Nn)
{
    int n = blockIdx.x * blockDim.x + threadIdx.x;
    if (n < Nn) cur[n] = xf[(size_t)n * TIN + (TIN - 1)];
}

// ---------------------------------------------------------------------------
extern "C" void kernel_launch(void* const* d_in, const int* in_sizes, int n_in,
                              void* d_out, int out_size, void* d_ws, size_t ws_size,
                              hipStream_t stream)
{
    const float* x      = (const float*)d_in[0];
    const int*   eidx   = (const int*)d_in[1];
    const float* eattr  = (const float*)d_in[2];
    const float* fus_fc1_w = (const float*)d_in[5];
    const float* fus_fc1_b = (const float*)d_in[6];
    const float* fus_fc2_w = (const float*)d_in[7];
    const float* fus_fc2_b = (const float*)d_in[8];
    const float* fus_g1_w  = (const float*)d_in[9];
    const float* fus_g1_b  = (const float*)d_in[10];
    const float* fus_g2_w  = (const float*)d_in[11];
    const float* fus_g2_b  = (const float*)d_in[12];
    const float* gnn_lin_w = (const float*)d_in[13];
    const float* gnn_lin_b = (const float*)d_in[14];
    const float* conv1_w   = (const float*)d_in[15];
    const float* conv1_b   = (const float*)d_in[16];
    const float* conv2_w   = (const float*)d_in[17];
    const float* conv2_b   = (const float*)d_in[18];
    const float* gru_wih   = (const float*)d_in[19];
    const float* gru_whh   = (const float*)d_in[20];
    const float* gru_bih   = (const float*)d_in[21];
    const float* gru_bhh   = (const float*)d_in[22];
    const float* pred_l1_w = (const float*)d_in[23];
    const float* pred_l1_b = (const float*)d_in[24];
    const float* ln1_g     = (const float*)d_in[25];
    const float* ln1_b     = (const float*)d_in[26];
    const float* pred_l2_w = (const float*)d_in[27];
    const float* pred_l2_b = (const float*)d_in[28];
    const float* ln2_g     = (const float*)d_in[29];
    const float* ln2_b     = (const float*)d_in[30];
    const float* pred_l3_w = (const float*)d_in[31];
    const float* pred_l3_b = (const float*)d_in[32];

    const int N = in_sizes[0] / TIN;
    const int E = in_sizes[1] / 2;
    const int G = in_sizes[3];
    const int* src = eidx;
    const int* dst = eidx + E;

    // ---- workspace arena (everything L2-resident: ~100 MB total) ----
    char* p = (char*)d_ws;
    auto alloc = [&](size_t bytes) -> char* {
        char* r = p;
        p += (bytes + 255) & ~(size_t)255;
        return r;
    };
    int*      me   = (int*)alloc((size_t)N * 4);
    int*      fo   = (int*)alloc((size_t)N * 4);
    float*    fdx  = (float*)alloc((size_t)N * 4);
    float*    deg  = (float*)alloc((size_t)N * 4);
    float*    dinv = (float*)alloc((size_t)N * 4);
    float*    xf   = (float*)alloc((size_t)N * TIN * 4);
    float*    h    = (float*)alloc((size_t)N * HID * 4);
    _Float16* hf   = (_Float16*)alloc((size_t)N * HID * 2);
    _Float16* af   = (_Float16*)alloc((size_t)N * HID * 2);
    float*    xw   = (float*)alloc((size_t)N * HID * 4);
    float*    agg  = (float*)alloc((size_t)N * HID * 4);
    float*    gi   = (float*)alloc((size_t)N * 3 * HID * 4);
    float*    gh   = (float*)alloc((size_t)N * 3 * HID * 4);
    float*    cur  = (float*)alloc((size_t)N * 4);
    _Float16* c1t  = (_Float16*)alloc((size_t)HID * HID * 2);
    _Float16* c2t  = (_Float16*)alloc((size_t)HID * HID * 2);
    _Float16* wiht = (_Float16*)alloc((size_t)HID * 3 * HID * 2);
    _Float16* whht = (_Float16*)alloc((size_t)HID * 3 * HID * 2);
    _Float16* l1t  = (_Float16*)alloc((size_t)HID * HID * 2);
    _Float16* l2t  = (_Float16*)alloc((size_t)HID * HID * 2);
    (void)ws_size; (void)n_in;

    const int B  = 256;
    const int nh = N * HID;
    auto gb = [&](long n) { return dim3((unsigned)((n + B - 1) / B)); };

    auto gemm = [&](const _Float16* A, const _Float16* Bt, const float* bias,
                    float* C, int M, int K, int Kout) {
        int tilesM    = M / 16;
        int rowBlocks = (tilesM + 7) / 8;       // 8 waves (row tiles) per block
        int colGroups = Kout / 64;              // 64 cols per block
        k_gemm_wmma<<<dim3(rowBlocks, colGroups), dim3(B), 0, stream>>>(
            A, Bt, bias, C, M, K, Kout);
    };

    // ---- one-time prep ----
    k_init_nd<<<gb(N), B, 0, stream>>>(me, deg, N);
    k_edge_prep<<<gb(E), B, 0, stream>>>(src, dst, me, deg, E);
    k_node_prep<<<gb(N), B, 0, stream>>>(me, dst, eattr, deg, fo, fdx, dinv, N, E);
    k_copy<<<gb((long)N * TIN), B, 0, stream>>>(x, xf, N * TIN);
    k_ghost<<<gb((long)G * TIN), B, 0, stream>>>(x, fo, fdx,
        fus_fc1_w, fus_fc1_b, fus_fc2_w, fus_fc2_b,
        fus_g1_w, fus_g1_b, fus_g2_w, fus_g2_b, xf, G);
    k_zero_h<<<gb(nh), B, 0, stream>>>(h, hf, nh);
    k_w2f16t<<<gb(HID * HID), B, 0, stream>>>(conv1_w, c1t, HID, HID);
    k_w2f16t<<<gb(HID * HID), B, 0, stream>>>(conv2_w, c2t, HID, HID);
    k_w2f16t<<<gb(HID * 3 * HID), B, 0, stream>>>(gru_wih, wiht, HID, 3 * HID);
    k_w2f16t<<<gb(HID * 3 * HID), B, 0, stream>>>(gru_whh, whht, HID, 3 * HID);
    k_w2f16t<<<gb(HID * HID), B, 0, stream>>>(pred_l1_w, l1t, HID, HID);
    k_w2f16t<<<gb(HID * HID), B, 0, stream>>>(pred_l2_w, l2t, HID, HID);

    // ---- gnn + gru pipeline for one timestep ----
    auto gnn_gru = [&](const float* xsrc, int stride) {
        k_lin_relu<<<gb(nh), B, 0, stream>>>(xsrc, stride, gnn_lin_w, gnn_lin_b, af, N);
        // conv1
        gemm(af, c1t, nullptr, xw, N, HID, HID);
        k_conv_init<<<gb(nh), B, 0, stream>>>(xw, dinv, agg, N);
        k_conv_edges<<<gb((long)E * 32), B, 0, stream>>>(src, dst, dinv, xw, agg, E);
        k_conv_fin<<<gb(nh), B, 0, stream>>>(agg, conv1_b, af, N);
        // conv2
        gemm(af, c2t, nullptr, xw, N, HID, HID);
        k_conv_init<<<gb(nh), B, 0, stream>>>(xw, dinv, agg, N);
        k_conv_edges<<<gb((long)E * 32), B, 0, stream>>>(src, dst, dinv, xw, agg, E);
        k_conv_fin<<<gb(nh), B, 0, stream>>>(agg, conv2_b, af, N);
        // gru
        gemm(af, wiht, gru_bih, gi, N, HID, 3 * HID);
        gemm(hf, whht, gru_bhh, gh, N, HID, 3 * HID);
        k_gru<<<gb(nh), B, 0, stream>>>(gi, gh, h, hf, N);
    };

    // ---- encoder: 6 timesteps ----
    for (int t = 0; t < TIN; ++t)
        gnn_gru(xf + t, TIN);

    // ---- decoder: 3 autoregressive steps ----
    k_set_cur<<<gb(N), B, 0, stream>>>(xf, cur, N);
    for (int t = 0; t < TOUT; ++t) {
        gnn_gru(cur, 1);
        gemm(hf, l1t, pred_l1_b, xw, N, HID, HID);
        k_ln_relu_f16<<<gb((long)N * 32), B, 0, stream>>>(xw, ln1_g, ln1_b, af, N);
        gemm(af, l2t, pred_l2_b, xw, N, HID, HID);
        k_ln_relu_f16<<<gb((long)N * 32), B, 0, stream>>>(xw, ln2_g, ln2_b, af, N);
        k_pred_out<<<gb((long)N * 32), B, 0, stream>>>(af, pred_l3_w, pred_l3_b,
                                                       cur, (float*)d_out, N, G, t);
    }
    (void)out_size;
}
